// PoseSPDecoderKT_37649683317547
// MI455X (gfx1250) — matrix operand order repaired and checked by reading, not compiled, not run on hardware
//
#include <hip/hip_runtime.h>
#include <hip/hip_bf16.h>

typedef __bf16 bf16_t;
typedef bf16_t v16bf __attribute__((ext_vector_type(16)));
typedef bf16_t bf16x8 __attribute__((ext_vector_type(8)));
typedef float  v8f   __attribute__((ext_vector_type(8)));

#define B_DIM 1024
#define W_DIM 512
#define T_DIM 64

__device__ const int KPAR[21] = {0,0,0,1,2,3,4,5,6,7,8,9,9,9,12,13,14,16,17,18,19};

// ---------------- CDNA5 async global->LDS helpers (ASYNCcnt-tracked, no VGPR round trip)
__device__ inline void async_ld16(void* lds_ptr, const bf16_t* gptr) {
    unsigned loff = (unsigned)(size_t)lds_ptr;  // generic shared addr low 32 = LDS offset
    asm volatile("global_load_async_to_lds_b128 %0, %1, off"
                 :: "v"(loff), "v"(gptr)
                 : "memory");
}
__device__ inline void wait_async_le4() {
    asm volatile("s_wait_asynccnt 0x4" ::: "memory");
}
__device__ inline void wait_async_0() {
    asm volatile("s_wait_asynccnt 0x0" ::: "memory");
}

// ---------------------------------------------------------------- prep kernels
__global__ void cvt_f32_bf16_kernel(const float* __restrict__ src, bf16_t* __restrict__ dst, int n) {
    int i = blockIdx.x * 256 + threadIdx.x;
    if (i < n) dst[i] = (bf16_t)src[i];
}

// conv weight (O=512,C=512,K=3) f32 -> (tap, o, c) bf16  [B-matrix n-major rows]
__global__ void conv_w_prep_kernel(const float* __restrict__ src, bf16_t* __restrict__ dst) {
    int i = blockIdx.x * 256 + threadIdx.x;
    if (i >= 3 * 512 * 512) return;
    int tap = i / (512 * 512);
    int rem = i - tap * 512 * 512;
    int o = rem >> 9, c = rem & 511;
    dst[i] = (bf16_t)src[(o * 512 + c) * 3 + tap];
}

// pw1 (21,1024,512) f32 -> pw1T (21,512,1024) bf16 : pw1T[j][h][c] = pw1[j][c][h]
__global__ void pw1_prep_kernel(const float* __restrict__ src, bf16_t* __restrict__ dst) {
    int i = blockIdx.x * 256 + threadIdx.x;
    if (i >= 21 * 512 * 1024) return;
    int j = i / (512 * 1024);
    int rem = i - j * 512 * 1024;
    int h = rem >> 10, c = rem & 1023;
    dst[i] = (bf16_t)src[((size_t)j * 1024 + c) * 512 + h];
}

// linear interpolation along L (reference interp_linear), bf16 in/out
__global__ void interp_kernel(const bf16_t* __restrict__ src, bf16_t* __restrict__ dst,
                              int Lin, int Lout) {
    int i = blockIdx.x * 256 + threadIdx.x;
    int total = B_DIM * Lout * 512;
    if (i >= total) return;
    int w = i & 511;
    int rest = i >> 9;
    int jj = rest % Lout;
    int b  = rest / Lout;
    float c = (jj + 0.5f) * ((float)Lin / (float)Lout) - 0.5f;
    c = fminf(fmaxf(c, 0.0f), (float)(Lin - 1));
    int lo = (int)floorf(c);
    int hi = lo + 1 < Lin ? lo + 1 : Lin - 1;
    float wt = c - (float)lo;
    float v = (1.0f - wt) * (float)src[((size_t)b * Lin + lo) * 512 + w]
            +          wt  * (float)src[((size_t)b * Lin + hi) * 512 + w];
    dst[((size_t)b * Lout + jj) * 512 + w] = (bf16_t)v;
}

// ------------------------------------------------- WMMA fragment helpers (wave32)
__device__ inline v16bf load_a_frag(const bf16_t* row, int half) {
    // A 16x32: lane half 0 -> K 0..7 & 16..23 ; half 1 -> K 8..15 & 24..31
    bf16x8 lo = *(const bf16x8*)(row + 8 * half);
    bf16x8 hi = *(const bf16x8*)(row + 16 + 8 * half);
    v16bf a;
#pragma unroll
    for (int e = 0; e < 8; ++e) { a[e] = lo[e]; a[8 + e] = hi[e]; }
    return a;
}
__device__ inline v16bf load_b_frag(const bf16_t* nrow, int half) {
    // B 32x16 stored n-major: lane half 0 -> K 0..7 & 16..23 ; half 1 -> K 8..15 & 24..31
    bf16x8 lo = *(const bf16x8*)(nrow + 8 * half);
    bf16x8 hi = *(const bf16x8*)(nrow + 16 + 8 * half);
    v16bf b;
#pragma unroll
    for (int e = 0; e < 8; ++e) { b[e] = lo[e]; b[8 + e] = hi[e]; }
    return b;
}

// -------------------------------------------------------- conv-as-GEMM (3 taps)
// out[(b,j), o] = relu?( sum_{tap,c} src[b, clamp(j+tap-1), c] * wT[tap][o][c] + bias[o] )
// Double-buffered async global->LDS staging.
__global__ __launch_bounds__(256)
void conv_gemm_kernel(const bf16_t* __restrict__ src,   // (B, L, 512)
                      const bf16_t* __restrict__ wT,    // (3, 512, 512) [tap][o][c]
                      const float*  __restrict__ bias,  // (512)
                      bf16_t* __restrict__ dst,         // (B, L, 512)
                      int L, int relu) {
    __shared__ bf16_t lds_a[2][128][40];
    __shared__ bf16_t lds_b[2][128][40];

    const int tid  = threadIdx.x;
    const int lane = tid & 31, wid = tid >> 5;
    const int half = lane >> 4, l16 = lane & 15;
    const int mblk = blockIdx.x * 128;
    const int nblk = blockIdx.y * 128;
    const int wm0 = (wid >> 1) * 32;
    const int wn0 = (wid & 1) * 64;

    v8f acc[2][4];
#pragma unroll
    for (int mi = 0; mi < 2; ++mi)
#pragma unroll
        for (int ni = 0; ni < 4; ++ni)
#pragma unroll
            for (int e = 0; e < 8; ++e) acc[mi][ni][e] = 0.0f;

    const int arow = tid >> 1;     // 0..127 LDS row this thread fills
    const int aseg = tid & 1;      // 16-element segment
    const int m = mblk + arow;
    const int b = m / L;
    const int j = m - b * L;

    // per-tap clamped A row base pointers (plus this thread's segment)
    const bf16_t* aptr[3];
#pragma unroll
    for (int tap = 0; tap < 3; ++tap) {
        int p = j + tap - 1;
        p = p < 0 ? 0 : (p >= L ? L - 1 : p);
        aptr[tap] = src + ((size_t)(b * L + p) * 512) + aseg * 16;
    }
    const bf16_t* bptr = wT + ((size_t)(nblk + arow) * 512) + aseg * 16;

    const int NSTAGE = 48;  // 3 taps x 16 K-steps
    auto issue = [&](int s, int buf) {
        int tap = s >> 4;
        int k0 = (s & 15) << 5;
        const bf16_t* ap = aptr[tap] + k0;
        const bf16_t* bp = bptr + (size_t)tap * 512 * 512 + k0;
        async_ld16(&lds_a[buf][arow][aseg * 16],     ap);
        async_ld16(&lds_a[buf][arow][aseg * 16 + 8], ap + 8);
        async_ld16(&lds_b[buf][arow][aseg * 16],     bp);
        async_ld16(&lds_b[buf][arow][aseg * 16 + 8], bp + 8);
    };

    issue(0, 0);
    for (int s = 0; s < NSTAGE; ++s) {
        const int cur = s & 1;
        if (s + 1 < NSTAGE) {
            issue(s + 1, cur ^ 1);
            wait_async_le4();   // stage s's 4 ops retired; next stage in flight
        } else {
            wait_async_0();
        }
        __syncthreads();

        v16bf afr[2];
#pragma unroll
        for (int mi = 0; mi < 2; ++mi)
            afr[mi] = load_a_frag(&lds_a[cur][wm0 + mi * 16 + l16][0], half);
#pragma unroll
        for (int ni = 0; ni < 4; ++ni) {
            v16bf bfr = load_b_frag(&lds_b[cur][wn0 + ni * 16 + l16][0], half);
#pragma unroll
            for (int mi = 0; mi < 2; ++mi)
                acc[mi][ni] = __builtin_amdgcn_wmma_f32_16x16x32_bf16(
                    false, afr[mi], false, bfr, (short)0, acc[mi][ni], false, false);
        }
        __syncthreads();   // buffer may be overwritten two stages later
    }

#pragma unroll
    for (int mi = 0; mi < 2; ++mi)
#pragma unroll
        for (int ni = 0; ni < 4; ++ni) {
            int nc = nblk + wn0 + ni * 16 + l16;
            float bv = bias[nc];
#pragma unroll
            for (int v = 0; v < 8; ++v) {
                int mr = mblk + wm0 + mi * 16 + half * 8 + v;
                float x = acc[mi][ni][v] + bv;
                if (relu) x = fmaxf(x, 0.0f);
                dst[(size_t)mr * 512 + nc] = (bf16_t)x;
            }
        }
}

// ---------------------------------------------- token mixing: per-b 64x512x64 GEMM
__global__ __launch_bounds__(256)
void tokmix_kernel(const bf16_t* __restrict__ H1,     // (B, 64, 512)
                   const bf16_t* __restrict__ tokWb,  // (64, 64) bf16
                   const float*  __restrict__ tok_b,  // (64)
                   bf16_t* __restrict__ H2) {         // (B, 64, 512)
    __shared__ bf16_t la[64][72];    // [s][t]
    __shared__ bf16_t lb[128][72];   // [n][t]

    const int tid = threadIdx.x;
    const int lane = tid & 31, wid = tid >> 5;
    const int half = lane >> 4, l16 = lane & 15;
    const int b = blockIdx.x;
    const int nblk = blockIdx.y * 128;

    { // load tokW: 64 rows x 4 segs
        int s = tid >> 2, seg = tid & 3;
        bf16x8 v0 = *(const bf16x8*)(tokWb + s * 64 + seg * 16);
        bf16x8 v1 = *(const bf16x8*)(tokWb + s * 64 + seg * 16 + 8);
        *(bf16x8*)&la[s][seg * 16] = v0;
        *(bf16x8*)&la[s][seg * 16 + 8] = v1;
    }
    // load H1 tile transposed: 64 t x 128 n
    for (int it = 0; it < 2; ++it) {
        int task = tid + 256 * it;
        int t = task >> 3, seg = task & 7;
        const bf16_t* p = H1 + ((size_t)(b * 64 + t) * 512) + nblk + seg * 16;
        bf16x8 v0 = *(const bf16x8*)(p);
        bf16x8 v1 = *(const bf16x8*)(p + 8);
#pragma unroll
        for (int q = 0; q < 8; ++q) {
            lb[seg * 16 + q][t]     = v0[q];
            lb[seg * 16 + 8 + q][t] = v1[q];
        }
    }
    __syncthreads();

    const int wm0 = (wid >> 2) * 32;   // 0 or 32
    const int wn0 = (wid & 3) * 32;    // 0,32,64,96
    v8f acc[2][2];
#pragma unroll
    for (int mi = 0; mi < 2; ++mi)
#pragma unroll
        for (int ni = 0; ni < 2; ++ni)
#pragma unroll
            for (int e = 0; e < 8; ++e) acc[mi][ni][e] = 0.0f;

#pragma unroll
    for (int kk = 0; kk < 64; kk += 32) {
        v16bf afr[2];
#pragma unroll
        for (int mi = 0; mi < 2; ++mi)
            afr[mi] = load_a_frag(&la[wm0 + mi * 16 + l16][kk], half);
#pragma unroll
        for (int ni = 0; ni < 2; ++ni) {
            v16bf bfr = load_b_frag(&lb[wn0 + ni * 16 + l16][kk], half);
#pragma unroll
            for (int mi = 0; mi < 2; ++mi)
                acc[mi][ni] = __builtin_amdgcn_wmma_f32_16x16x32_bf16(
                    false, afr[mi], false, bfr, (short)0, acc[mi][ni], false, false);
        }
    }
#pragma unroll
    for (int mi = 0; mi < 2; ++mi)
#pragma unroll
        for (int ni = 0; ni < 2; ++ni) {
            int nc = nblk + wn0 + ni * 16 + l16;
#pragma unroll
            for (int v = 0; v < 8; ++v) {
                int s = wm0 + mi * 16 + half * 8 + v;
                float x = fmaxf(acc[mi][ni][v] + tok_b[s], 0.0f);
                H2[((size_t)(b * 64 + s) * 512) + nc] = (bf16_t)x;
            }
        }
}

// ------------------------------------- per-joint FC1: (1024 x 512 x 1024) x 21
// A rows gather (curr || parent) features; double-buffered async staging.
__global__ __launch_bounds__(256)
void fc1_kernel(const bf16_t* __restrict__ h22,    // (B, 22, 512)
                const bf16_t* __restrict__ pw1T,   // (21, 512, 1024) [j][h][c]
                const float*  __restrict__ pb1,    // (21, 512)
                bf16_t* __restrict__ hid) {        // (21, B, 512)
    __shared__ bf16_t lds_a[2][128][40];
    __shared__ bf16_t lds_b[2][128][40];

    const int tid = threadIdx.x;
    const int lane = tid & 31, wid = tid >> 5;
    const int half = lane >> 4, l16 = lane & 15;
    const int j = blockIdx.z;
    const int mblk = blockIdx.x * 128;
    const int nblk = blockIdx.y * 128;
    const int wm0 = (wid >> 1) * 32;
    const int wn0 = (wid & 1) * 64;
    const int curr = j + 1;
    const int par  = KPAR[j];

    v8f acc[2][4];
#pragma unroll
    for (int mi = 0; mi < 2; ++mi)
#pragma unroll
        for (int ni = 0; ni < 4; ++ni)
#pragma unroll
            for (int e = 0; e < 8; ++e) acc[mi][ni][e] = 0.0f;

    const int arow = tid >> 1;
    const int aseg = tid & 1;
    const int bidx = mblk + arow;

    const bf16_t* apc = h22 + ((size_t)bidx * 22 + curr) * 512 + aseg * 16;
    const bf16_t* app = h22 + ((size_t)bidx * 22 + par ) * 512 + aseg * 16;
    const bf16_t* bpb = pw1T + ((size_t)j * 512 + nblk + arow) * 1024 + aseg * 16;

    const int NSTAGE = 32;  // K = 1024 in steps of 32
    auto issue = [&](int s, int buf) {
        int k0 = s << 5;
        const bf16_t* ap = ((k0 >> 9) ? app : apc) + (k0 & 511);
        const bf16_t* bp = bpb + k0;
        async_ld16(&lds_a[buf][arow][aseg * 16],     ap);
        async_ld16(&lds_a[buf][arow][aseg * 16 + 8], ap + 8);
        async_ld16(&lds_b[buf][arow][aseg * 16],     bp);
        async_ld16(&lds_b[buf][arow][aseg * 16 + 8], bp + 8);
    };

    issue(0, 0);
    for (int s = 0; s < NSTAGE; ++s) {
        const int cur = s & 1;
        if (s + 1 < NSTAGE) {
            issue(s + 1, cur ^ 1);
            wait_async_le4();
        } else {
            wait_async_0();
        }
        __syncthreads();

        v16bf afr[2];
#pragma unroll
        for (int mi = 0; mi < 2; ++mi)
            afr[mi] = load_a_frag(&lds_a[cur][wm0 + mi * 16 + l16][0], half);
#pragma unroll
        for (int ni = 0; ni < 4; ++ni) {
            v16bf bfr = load_b_frag(&lds_b[cur][wn0 + ni * 16 + l16][0], half);
#pragma unroll
            for (int mi = 0; mi < 2; ++mi)
                acc[mi][ni] = __builtin_amdgcn_wmma_f32_16x16x32_bf16(
                    false, afr[mi], false, bfr, (short)0, acc[mi][ni], false, false);
        }
        __syncthreads();
    }

#pragma unroll
    for (int mi = 0; mi < 2; ++mi)
#pragma unroll
        for (int ni = 0; ni < 4; ++ni) {
            int nc = nblk + wn0 + ni * 16 + l16;
            float bv = pb1[j * 512 + nc];
#pragma unroll
            for (int v = 0; v < 8; ++v) {
                int mr = mblk + wm0 + mi * 16 + half * 8 + v;
                float x = fmaxf(acc[mi][ni][v] + bv, 0.0f);
                hid[((size_t)j * 1024 + mr) * 512 + nc] = (bf16_t)x;
            }
        }
}

// ----------------------------------------- layernorm head -> pred_beta (B,10)
__global__ __launch_bounds__(256)
void beta_kernel(const bf16_t* __restrict__ h22, const float* __restrict__ ln_g,
                 const float* __restrict__ ln_b, const float* __restrict__ beta_w,
                 const float* __restrict__ beta_b, float* __restrict__ out) {
    __shared__ float red1[256], red2[256], accb[10];
    const int tid = threadIdx.x;
    const int b = blockIdx.x;
    const int c0 = tid * 2;
    float m0 = 0.f, m1 = 0.f;
    for (int j = 0; j < 22; ++j) {
        const bf16_t* r = h22 + ((size_t)b * 22 + j) * 512;
        m0 += (float)r[c0];
        m1 += (float)r[c0 + 1];
    }
    m0 *= (1.0f / 22.0f); m1 *= (1.0f / 22.0f);
    red1[tid] = m0 + m1;
    red2[tid] = m0 * m0 + m1 * m1;
    if (tid < 10) accb[tid] = 0.f;
    __syncthreads();
    for (int s = 128; s > 0; s >>= 1) {
        if (tid < s) { red1[tid] += red1[tid + s]; red2[tid] += red2[tid + s]; }
        __syncthreads();
    }
    float mu = red1[0] * (1.0f / 512.0f);
    float var = red2[0] * (1.0f / 512.0f) - mu * mu;
    float rstd = rsqrtf(var + 1e-5f);
    float n0 = (m0 - mu) * rstd * ln_g[c0]     + ln_b[c0];
    float n1 = (m1 - mu) * rstd * ln_g[c0 + 1] + ln_b[c0 + 1];
    float part[10];
#pragma unroll
    for (int o = 0; o < 10; ++o)
        part[o] = n0 * beta_w[c0 * 10 + o] + n1 * beta_w[(c0 + 1) * 10 + o];
#pragma unroll
    for (int o = 0; o < 10; ++o) atomicAdd(&accb[o], part[o]);
    __syncthreads();
    if (tid < 10) out[b * 10 + tid] = accb[tid] + beta_b[tid];
}

// --------------------------------- FC2 (512x6) + 6D -> rotation matrix, per (b,j)
__global__ __launch_bounds__(256)
void fc2rot_kernel(const bf16_t* __restrict__ hid,   // (21, B, 512)
                   const float* __restrict__ pw2,    // (21, 512, 6)
                   const float* __restrict__ pb2,    // (21, 6)
                   float* __restrict__ pose_out,     // (B, 21, 6)
                   float* __restrict__ rot_out) {    // (B, 21, 9)
    const int g = blockIdx.x * 8 + (threadIdx.x >> 5);
    const int lane = threadIdx.x & 31;
    if (g >= B_DIM * 21) return;
    const int b = g / 21, j = g - b * 21;
    const bf16_t* hp = hid + ((size_t)j * 1024 + b) * 512;
    const float* wp = pw2 + (size_t)j * 512 * 6;
    float acc[6] = {0, 0, 0, 0, 0, 0};
    for (int k = lane; k < 512; k += 32) {
        float h = (float)hp[k];
#pragma unroll
        for (int d = 0; d < 6; ++d) acc[d] += h * wp[k * 6 + d];
    }
#pragma unroll
    for (int off = 16; off > 0; off >>= 1)
#pragma unroll
        for (int d = 0; d < 6; ++d) acc[d] += __shfl_down(acc[d], off);
    if (lane == 0) {
        float p[6];
#pragma unroll
        for (int d = 0; d < 6; ++d) {
            p[d] = acc[d] + pb2[j * 6 + d];
            pose_out[(size_t)(b * 21 + j) * 6 + d] = p[d];
        }
        float a1[3] = {p[0], p[1], p[2]}, a2[3] = {p[3], p[4], p[5]};
        float inv = rsqrtf(a1[0]*a1[0] + a1[1]*a1[1] + a1[2]*a1[2]);
        float b1[3] = {a1[0]*inv, a1[1]*inv, a1[2]*inv};
        float dd = b1[0]*a2[0] + b1[1]*a2[1] + b1[2]*a2[2];
        float a2p[3] = {a2[0]-dd*b1[0], a2[1]-dd*b1[1], a2[2]-dd*b1[2]};
        float inv2 = rsqrtf(a2p[0]*a2p[0] + a2p[1]*a2p[1] + a2p[2]*a2p[2]);
        float b2[3] = {a2p[0]*inv2, a2p[1]*inv2, a2p[2]*inv2};
        float b3[3] = {b1[1]*b2[2]-b1[2]*b2[1], b1[2]*b2[0]-b1[0]*b2[2], b1[0]*b2[1]-b1[1]*b2[0]};
        float* r = rot_out + (size_t)(b * 21 + j) * 9;
#pragma unroll
        for (int c = 0; c < 3; ++c) { r[c] = b1[c]; r[3 + c] = b2[c]; r[6 + c] = b3[c]; }
    }
}

// ---------------------------------------------------------------------- launch
extern "C" void kernel_launch(void* const* d_in, const int* in_sizes, int n_in,
                              void* d_out, int out_size, void* d_ws, size_t ws_size,
                              hipStream_t stream) {
    const float* x       = (const float*)d_in[0];
    const float* conv1_w = (const float*)d_in[1];
    const float* conv1_b = (const float*)d_in[2];
    const float* tok_w   = (const float*)d_in[3];
    const float* tok_b   = (const float*)d_in[4];
    const float* up_w    = (const float*)d_in[5];
    const float* up_b    = (const float*)d_in[6];
    const float* ln_g    = (const float*)d_in[7];
    const float* ln_b    = (const float*)d_in[8];
    const float* beta_w  = (const float*)d_in[9];
    const float* beta_b  = (const float*)d_in[10];
    const float* pw1     = (const float*)d_in[11];
    const float* pb1     = (const float*)d_in[12];
    const float* pw2     = (const float*)d_in[13];
    const float* pb2     = (const float*)d_in[14];
    float* out = (float*)d_out;

    char* ws = (char*)d_ws;
    constexpr size_t SZ_CONVW = (size_t)3 * 512 * 512 * 2;          // 1.5 MB each
    constexpr size_t OFF_WC1  = 0;
    constexpr size_t OFF_WU0  = OFF_WC1 + SZ_CONVW;
    constexpr size_t OFF_WU1  = OFF_WU0 + SZ_CONVW;
    constexpr size_t OFF_WU2  = OFF_WU1 + SZ_CONVW;
    constexpr size_t OFF_TOKW = OFF_WU2 + SZ_CONVW;
    constexpr size_t OFF_PW1T = OFF_TOKW + 64 * 64 * 2;
    constexpr size_t OFF_S0   = OFF_PW1T + (size_t)21 * 512 * 1024 * 2;
    constexpr size_t SZ_SLOT  = (size_t)B_DIM * 64 * 512 * 2;       // 67 MB
    constexpr size_t OFF_S1   = OFF_S0 + SZ_SLOT;

    bf16_t* WC1  = (bf16_t*)(ws + OFF_WC1);
    bf16_t* WU0  = (bf16_t*)(ws + OFF_WU0);
    bf16_t* WU1  = (bf16_t*)(ws + OFF_WU1);
    bf16_t* WU2  = (bf16_t*)(ws + OFF_WU2);
    bf16_t* TOKW = (bf16_t*)(ws + OFF_TOKW);
    bf16_t* PW1T = (bf16_t*)(ws + OFF_PW1T);
    bf16_t* S0   = (bf16_t*)(ws + OFF_S0);
    bf16_t* S1   = (bf16_t*)(ws + OFF_S1);

    // prep
    cvt_f32_bf16_kernel<<<(B_DIM * 64 * 512 + 255) / 256, 256, 0, stream>>>(x, S0, B_DIM * 64 * 512);
    cvt_f32_bf16_kernel<<<(64 * 64 + 255) / 256, 256, 0, stream>>>(tok_w, TOKW, 64 * 64);
    conv_w_prep_kernel<<<(3 * 512 * 512 + 255) / 256, 256, 0, stream>>>(conv1_w, WC1);
    conv_w_prep_kernel<<<(3 * 512 * 512 + 255) / 256, 256, 0, stream>>>(up_w + 0 * 512 * 512 * 3, WU0);
    conv_w_prep_kernel<<<(3 * 512 * 512 + 255) / 256, 256, 0, stream>>>(up_w + 1 * 512 * 512 * 3, WU1);
    conv_w_prep_kernel<<<(3 * 512 * 512 + 255) / 256, 256, 0, stream>>>(up_w + 2 * 512 * 512 * 3, WU2);
    pw1_prep_kernel<<<(21 * 512 * 1024 + 255) / 256, 256, 0, stream>>>(pw1, PW1T);

    // conv1 (L=64, relu) : S0 -> S1
    conv_gemm_kernel<<<dim3(512, 4), 256, 0, stream>>>(S0, WC1, conv1_b, S1, 64, 1);
    // token mix (relu) : S1 -> S0
    tokmix_kernel<<<dim3(B_DIM, 4), 256, 0, stream>>>(S1, TOKW, tok_b, S0);
    // interp 64->50 : S0 -> S1 ; conv up0 (relu) : S1 -> S0
    interp_kernel<<<(B_DIM * 50 * 512 + 255) / 256, 256, 0, stream>>>(S0, S1, 64, 50);
    conv_gemm_kernel<<<dim3(400, 4), 256, 0, stream>>>(S1, WU0, up_b + 0 * 512, S0, 50, 1);
    // interp 50->36 : S0 -> S1 ; conv up1 (relu) : S1 -> S0
    interp_kernel<<<(B_DIM * 36 * 512 + 255) / 256, 256, 0, stream>>>(S0, S1, 50, 36);
    conv_gemm_kernel<<<dim3(288, 4), 256, 0, stream>>>(S1, WU1, up_b + 1 * 512, S0, 36, 1);
    // interp 36->22 : S0 -> S1 ; conv up2 (NO relu) : S1 -> S0  (= h22)
    interp_kernel<<<(B_DIM * 22 * 512 + 255) / 256, 256, 0, stream>>>(S0, S1, 36, 22);
    conv_gemm_kernel<<<dim3(176, 4), 256, 0, stream>>>(S1, WU2, up_b + 2 * 512, S0, 22, 0);

    // heads
    beta_kernel<<<B_DIM, 256, 0, stream>>>(S0, ln_g, ln_b, beta_w, beta_b, out + 129024 + 193536);
    fc1_kernel<<<dim3(8, 4, 21), 256, 0, stream>>>(S0, PW1T, pb1, S1);
    fc2rot_kernel<<<(B_DIM * 21) / 8, 256, 0, stream>>>(S1, pw2, pb2, out, out + 129024);

    (void)in_sizes; (void)n_in; (void)out_size; (void)ws_size;
}
